// FastestFlashGQA_738734374970
// MI455X (gfx1250) — matrix operand order repaired and checked by reading, not compiled, not run on hardware
//
#include <hip/hip_runtime.h>
#include <hip/hip_bf16.h>
#include <stdint.h>

typedef __bf16 bf16;
typedef __attribute__((ext_vector_type(16))) __bf16 v16bf;
typedef __attribute__((ext_vector_type(8)))  __bf16 v8bf;
typedef __attribute__((ext_vector_type(8)))  float  v8f;

#define B_      2
#define S_      2048
#define HID_    1024
#define NH_     16
#define NKV_    4
#define D_      64
#define WINDOW_ 1024
#define SCALE_  0.125f
#define SOFTCAP_ 30.0f
#define NEGBIG_ -1.0e30f

// ---------------------------------------------------------------------------
// WMMA helpers (CDNA5 v_wmma_f32_16x16x32_bf16, wave32)
// ---------------------------------------------------------------------------
static __device__ __forceinline__ v8f wmma_bf16(v16bf a, v16bf b, v8f c) {
  // (neg_a, A, neg_b, B, c_mod, C, reuse_a, reuse_b)
  return __builtin_amdgcn_wmma_f32_16x16x32_bf16(false, a, false, b, (short)0, c,
                                                 false, false);
}

// A-matrix fragment (16x32, row-major source, ld in elements).
// Layout used here (contiguous k, matches load_b_frag): lane L holds row M=L&15; elements 0..15 = K = (L>>4)*16 + 0..15
// -> two adjacent 16B chunks.
static __device__ __forceinline__ v16bf load_a_frag(const bf16* base, int ld, int kb) {
  int lane = (int)(threadIdx.x & 31);
  int half = lane >> 4;
  int row  = lane & 15;
  const bf16* p = base + (size_t)row * ld + kb + half * 16;
  union { v16bf v; v8bf h[2]; } u;
  u.h[0] = *(const v8bf*)(p);
  u.h[1] = *(const v8bf*)(p + 8);
  return u.v;
}

// B-matrix fragment (32x16). Source is [N][K] row-major (ld in elements),
// i.e. B[k][n] = src[n][k].  Layout used here (contiguous k): lane L holds column N=L&15;
// elements 0..15 = K = (L>>4)*16 + 0..15 -> one contiguous 32B chunk.
static __device__ __forceinline__ v16bf load_b_frag(const bf16* base, int ld, int kb) {
  int lane = (int)(threadIdx.x & 31);
  int half = lane >> 4;
  int col  = lane & 15;
  const bf16* p = base + (size_t)col * ld + kb + half * 16;
  return *(const v16bf*)(p);
}

// xor-shuffle via LDS-permute hardware (no HIP warp-size pitfalls).
static __device__ __forceinline__ float lane_xor_f(float v, int m) {
  int lane = (int)(threadIdx.x & 31);
  int r = __builtin_amdgcn_ds_bpermute((lane ^ m) << 2, __builtin_bit_cast(int, v));
  return __builtin_bit_cast(float, r);
}

// ---------------------------------------------------------------------------
// f32 -> bf16 convert
// ---------------------------------------------------------------------------
__global__ void cvt_f32_bf16(const float* __restrict__ in, bf16* __restrict__ out, int n) {
  int i = blockIdx.x * blockDim.x + threadIdx.x;
  int stride = gridDim.x * blockDim.x;
  for (; i < n; i += stride) out[i] = (bf16)in[i];
}

// ---------------------------------------------------------------------------
// Generic GEMM: C[M,N] = A[M,K](bf16) * W[N,K](bf16)^T
// One wave computes a 16x64 tile.  mode: 0 = bf16 row-major out,
// 1 = bf16 out scattered as Vt[(b*NKV+kvh)*64+dim][2048 tokens],
// 2 = f32 row-major out.
// ---------------------------------------------------------------------------
__global__ __launch_bounds__(256) void gemm_bf16_wmma(
    const bf16* __restrict__ A, int lda,
    const bf16* __restrict__ W, int ldb,
    void* __restrict__ C, int M, int N, int K, int mode)
{
  int w = (int)(threadIdx.x >> 5);
  int waveId = blockIdx.x * (blockDim.x >> 5) + w;
  int wavesPerN = N >> 6;
  int mBase = (waveId / wavesPerN) * 16;
  int nBase = (waveId % wavesPerN) * 64;
  if (mBase >= M) return;

  v8f acc[4];
#pragma unroll
  for (int t = 0; t < 4; ++t)
#pragma unroll
    for (int g = 0; g < 8; ++g) acc[t][g] = 0.f;

  const bf16* Arow = A + (size_t)mBase * lda;
  for (int kb = 0; kb < K; kb += 32) {
    v16bf a = load_a_frag(Arow, lda, kb);
#pragma unroll
    for (int t = 0; t < 4; ++t) {
      v16bf b = load_b_frag(W + (size_t)(nBase + t * 16) * ldb, ldb, kb);
      acc[t] = wmma_bf16(a, b, acc[t]);
    }
  }

  int lane = (int)(threadIdx.x & 31);
  int hi = lane >> 4, nn = lane & 15;
#pragma unroll
  for (int t = 0; t < 4; ++t) {
#pragma unroll
    for (int g = 0; g < 8; ++g) {
      int m = mBase + hi * 8 + g;
      int n = nBase + t * 16 + nn;
      float v = acc[t][g];
      if (mode == 0) {
        ((bf16*)C)[(size_t)m * N + n] = (bf16)v;
      } else if (mode == 1) {
        int bb = m >> 11, s = m & (S_ - 1);
        int kvh = n >> 6, dim = n & (D_ - 1);
        ((bf16*)C)[(((size_t)bb * NKV_ + kvh) * D_ + dim) * S_ + s] = (bf16)v;
      } else {
        ((float*)C)[(size_t)m * N + n] = v;
      }
    }
  }
}

// ---------------------------------------------------------------------------
// Flash attention, one wave per (b, h, 16-query tile).
// Q: [B*S, NH*D] bf16 ; Kmat: [B*S, NKV*D] bf16 ; Vt: [B*NKV*D, S] bf16
// O: [B*S, NH*D] bf16
// ---------------------------------------------------------------------------
__global__ __launch_bounds__(256) void attn_wmma(
    const bf16* __restrict__ Q, const bf16* __restrict__ Kmat,
    const bf16* __restrict__ Vt, bf16* __restrict__ O)
{
  __shared__ bf16 Plds[8][16 * 32];

  int lane = (int)(threadIdx.x & 31);
  int w = (int)(threadIdx.x >> 5);
  int waveId = blockIdx.x * (blockDim.x >> 5) + w;
  int qt = waveId & (S_ / 16 - 1);         // 128 query tiles per (b,h)
  int h  = (waveId >> 7) & (NH_ - 1);
  int b  = waveId >> 11;
  if (b >= B_) return;
  int kvh = h >> 2;                        // n_rep = 4, consecutive repeat
  int i0 = qt * 16;
  int hi = lane >> 4, nn = lane & 15;

  const bf16* Qbase = Q + ((size_t)(b * S_ + i0)) * (NH_ * D_) + h * D_;
  v16bf aQ0 = load_a_frag(Qbase, NH_ * D_, 0);
  v16bf aQ1 = load_a_frag(Qbase, NH_ * D_, 32);

  float mrow[8], lrow[8];
  v8f o[4];
#pragma unroll
  for (int g = 0; g < 8; ++g) { mrow[g] = NEGBIG_; lrow[g] = 0.f; }
#pragma unroll
  for (int t = 0; t < 4; ++t)
#pragma unroll
    for (int g = 0; g < 8; ++g) o[t][g] = 0.f;

  int jLo = i0 - (WINDOW_ - 1); if (jLo < 0) jLo = 0;
  const bf16* Vbase = Vt + ((size_t)(b * NKV_ + kvh) * D_) * S_;
  bf16* P = Plds[w];

  for (int j0 = jLo & ~31; j0 <= i0 + 15; j0 += 32) {
    const bf16* Kb0 = Kmat + ((size_t)(b * S_ + j0)) * (NKV_ * D_) + kvh * D_;
    v8f c0, c1;
#pragma unroll
    for (int g = 0; g < 8; ++g) { c0[g] = 0.f; c1[g] = 0.f; }
    c0 = wmma_bf16(aQ0, load_b_frag(Kb0, NKV_ * D_, 0),  c0);
    c0 = wmma_bf16(aQ1, load_b_frag(Kb0, NKV_ * D_, 32), c0);
    bool do1 = (j0 + 16 <= i0 + 15);       // uniform across the wave
    if (do1) {
      const bf16* Kb1 = Kb0 + (size_t)16 * (NKV_ * D_);
      c1 = wmma_bf16(aQ0, load_b_frag(Kb1, NKV_ * D_, 0),  c1);
      c1 = wmma_bf16(aQ1, load_b_frag(Kb1, NKV_ * D_, 32), c1);
    }

    float e0[8], e1[8], alpha[8];
#pragma unroll
    for (int g = 0; g < 8; ++g) {
      int qi = i0 + hi * 8 + g;
      int j  = j0 + nn;
      float s0 = SOFTCAP_ * tanhf(c0[g] * (SCALE_ / SOFTCAP_));
      float s1 = SOFTCAP_ * tanhf(c1[g] * (SCALE_ / SOFTCAP_));
      bool ok0 = (j <= qi) && (j > qi - WINDOW_);
      bool ok1 = do1 && ((j + 16) <= qi) && ((j + 16) > qi - WINDOW_);
      s0 = ok0 ? s0 : NEGBIG_;
      s1 = ok1 ? s1 : NEGBIG_;
      float tmax = fmaxf(s0, s1);
      tmax = fmaxf(tmax, lane_xor_f(tmax, 1));
      tmax = fmaxf(tmax, lane_xor_f(tmax, 2));
      tmax = fmaxf(tmax, lane_xor_f(tmax, 4));
      tmax = fmaxf(tmax, lane_xor_f(tmax, 8));
      float mnew = fmaxf(mrow[g], tmax);
      alpha[g] = __expf(mrow[g] - mnew);
      mrow[g]  = mnew;
      e0[g] = (s0 > -1.0e29f) ? __expf(s0 - mnew) : 0.f;
      e1[g] = (s1 > -1.0e29f) ? __expf(s1 - mnew) : 0.f;
      float r = e0[g] + e1[g];
      r += lane_xor_f(r, 1);
      r += lane_xor_f(r, 2);
      r += lane_xor_f(r, 4);
      r += lane_xor_f(r, 8);
      lrow[g] = lrow[g] * alpha[g] + r;
    }

    // C-layout -> A-layout via per-wave LDS tile (bf16 [16][32])
#pragma unroll
    for (int g = 0; g < 8; ++g) {
      int r = hi * 8 + g;
      P[r * 32 + nn]      = (bf16)e0[g];
      P[r * 32 + 16 + nn] = (bf16)e1[g];
#pragma unroll
      for (int t = 0; t < 4; ++t) o[t][g] *= alpha[g];
    }
    asm volatile("s_wait_dscnt 0" ::: "memory");
    v16bf aP = load_a_frag(P, 32, 0);
#pragma unroll
    for (int t = 0; t < 4; ++t) {
      const bf16* Vp = Vbase + (size_t)(t * 16 + nn) * S_ + j0 + hi * 16;
      v16bf bV = *(const v16bf*)Vp;     // B-fragment directly from Vt
      o[t] = wmma_bf16(aP, bV, o[t]);
    }
  }

#pragma unroll
  for (int g = 0; g < 8; ++g) {
    float inv = 1.0f / lrow[g];
    int qi = i0 + hi * 8 + g;
    size_t rowoff = ((size_t)(b * S_ + qi)) * (NH_ * D_) + h * D_;
#pragma unroll
    for (int t = 0; t < 4; ++t)
      O[rowoff + t * 16 + nn] = (bf16)(o[t][g] * inv);
  }
}

// ---------------------------------------------------------------------------
extern "C" void kernel_launch(void* const* d_in, const int* in_sizes, int n_in,
                              void* d_out, int out_size, void* d_ws, size_t ws_size,
                              hipStream_t stream) {
  const float* x  = (const float*)d_in[0];
  const float* Wq = (const float*)d_in[1];
  const float* Wk = (const float*)d_in[2];
  const float* Wv = (const float*)d_in[3];
  const float* Wo = (const float*)d_in[4];
  float* out = (float*)d_out;

  const int M = B_ * S_;               // 4096 tokens
  char* ws = (char*)d_ws;
  size_t off = 0;
  auto alloc = [&](size_t bytes) -> void* {
    void* p = ws + off;
    off += (bytes + 255) & ~(size_t)255;
    return p;
  };
  bf16* Xb  = (bf16*)alloc((size_t)M * HID_ * 2);
  bf16* Wqb = (bf16*)alloc((size_t)(NH_ * D_) * HID_ * 2);
  bf16* Wkb = (bf16*)alloc((size_t)(NKV_ * D_) * HID_ * 2);
  bf16* Wvb = (bf16*)alloc((size_t)(NKV_ * D_) * HID_ * 2);
  bf16* Wob = (bf16*)alloc((size_t)HID_ * (NH_ * D_) * 2);
  bf16* Qb  = (bf16*)alloc((size_t)M * (NH_ * D_) * 2);
  bf16* Kb  = (bf16*)alloc((size_t)M * (NKV_ * D_) * 2);
  bf16* Vtb = (bf16*)alloc((size_t)M * (NKV_ * D_) * 2 + 4096); // pad: tile overread
  bf16* Ob  = (bf16*)alloc((size_t)M * (NH_ * D_) * 2);

  dim3 blk(256);
  cvt_f32_bf16<<<2048, blk, 0, stream>>>(x,  Xb,  M * HID_);
  cvt_f32_bf16<<<1024, blk, 0, stream>>>(Wq, Wqb, NH_ * D_ * HID_);
  cvt_f32_bf16<<<512,  blk, 0, stream>>>(Wk, Wkb, NKV_ * D_ * HID_);
  cvt_f32_bf16<<<512,  blk, 0, stream>>>(Wv, Wvb, NKV_ * D_ * HID_);
  cvt_f32_bf16<<<1024, blk, 0, stream>>>(Wo, Wob, HID_ * NH_ * D_);

  { // Q = Xb * Wq^T  -> bf16 [M, 1024]
    int waves = (M / 16) * ((NH_ * D_) / 64);
    gemm_bf16_wmma<<<waves / 8, blk, 0, stream>>>(Xb, HID_, Wqb, HID_, Qb,
                                                  M, NH_ * D_, HID_, 0);
  }
  { // K = Xb * Wk^T  -> bf16 [M, 256]
    int waves = (M / 16) * ((NKV_ * D_) / 64);
    gemm_bf16_wmma<<<waves / 8, blk, 0, stream>>>(Xb, HID_, Wkb, HID_, Kb,
                                                  M, NKV_ * D_, HID_, 0);
  }
  { // V = Xb * Wv^T  -> bf16 transposed Vt[(b,kvh,dim), s]
    int waves = (M / 16) * ((NKV_ * D_) / 64);
    gemm_bf16_wmma<<<waves / 8, blk, 0, stream>>>(Xb, HID_, Wvb, HID_, Vtb,
                                                  M, NKV_ * D_, HID_, 1);
  }
  { // flash attention
    int waves = B_ * NH_ * (S_ / 16);
    attn_wmma<<<waves / 8, blk, 0, stream>>>(Qb, Kb, Vtb, Ob);
  }
  { // out = O * Wo^T -> f32 [M, 1024]
    int waves = (M / 16) * (HID_ / 64);
    gemm_bf16_wmma<<<waves / 8, blk, 0, stream>>>(Ob, NH_ * D_, Wob, NH_ * D_,
                                                  out, M, HID_, NH_ * D_, 2);
  }
}
